// HeterogeneousMultiTaskGNN_50800873177107
// MI455X (gfx1250) — matrix operand order repaired and checked by reading, not compiled, not run on hardware
//
#include <hip/hip_runtime.h>
#include <hip/hip_bf16.h>

typedef __attribute__((ext_vector_type(16))) _Float16 v16h;
typedef __attribute__((ext_vector_type(8)))  _Float16 v8h;
typedef __attribute__((ext_vector_type(8)))  float    v8f;

#define DEVINL __device__ __forceinline__

union AV16 { v16h v; v8h h[2]; };

// ---- float atomic max: prefer native GLOBAL_ATOMIC_MAX_NUM_F32 via HIP's
// float atomicMax overload; fall back to a CAS loop if the overload is absent.
DEVINL void atomicMaxF_cas(float* addr, float val) {
  unsigned int* ua = (unsigned int*)addr;
  unsigned int cur = *(volatile unsigned int*)ua;
  while (true) {
    float f = __uint_as_float(cur);
    if (f >= val) break;
    unsigned int assumed = cur;
    cur = atomicCAS(ua, assumed, __float_as_uint(val));
    if (cur == assumed) break;
  }
}
template <typename T>
DEVINL auto atomicMaxF_impl(T* addr, T val, int) -> decltype(atomicMax(addr, val), void()) {
  atomicMax(addr, val);                       // native float max atomic
}
template <typename T>
DEVINL void atomicMaxF_impl(T* addr, T val, long) {
  atomicMaxF_cas(addr, val);                  // CAS fallback
}
DEVINL void atomicMaxF(float* addr, float val) { atomicMaxF_impl(addr, val, 0); }

DEVINL float lrelu02(float x) { return x > 0.f ? x : 0.2f * x; }

// ---------------------------------------------------------------- utility
__global__ void k_fill(float* __restrict__ p, float v, long n) {
  long i = (long)blockIdx.x * blockDim.x + threadIdx.x;
  if (i < n) p[i] = v;
}

__global__ void k_cvt16(const float* __restrict__ s, _Float16* __restrict__ d, long n) {
  long i = (long)blockIdx.x * blockDim.x + threadIdx.x;
  if (i < n) d[i] = (_Float16)s[i];
}

// ------------------------------------------------- self-loop attr (mean of incoming)
__global__ void k_deg_loop(const long long* __restrict__ ei, const float* __restrict__ ea,
                           float* __restrict__ deg, float* __restrict__ lsum, long E) {
  long e = (long)blockIdx.x * blockDim.x + threadIdx.x;
  if (e >= E) return;
  long c = ei[E + e];
  atomicAdd(&deg[c], 1.f);
#pragma unroll
  for (int d = 0; d < 16; ++d) atomicAdd(&lsum[c * 16 + d], ea[e * 16 + d]);
}

__global__ void k_loop_fin(const float* __restrict__ deg, float* __restrict__ lsum, long N) {
  long i = (long)blockIdx.x * blockDim.x + threadIdx.x;
  if (i >= N * 16) return;
  long n = i >> 4;
  lsum[i] /= fmaxf(deg[n], 1.f);
}

__global__ void k_domcnt(const long long* __restrict__ dom, float* __restrict__ cnt, long N) {
  long i = (long)blockIdx.x * blockDim.x + threadIdx.x;
  if (i < N) atomicAdd(&cnt[(int)dom[i]], 1.f);
}

__global__ void k_projbias(const float* __restrict__ cnt, const float* __restrict__ demb,
                           const float* __restrict__ pb, float* __restrict__ bias, float invN) {
  int j = threadIdx.x;            // 64 threads
  if (j >= 64) return;
  float de = 0.f;
  if (j < 16) {
    for (int d = 0; d < 4; ++d) de += cnt[d] * demb[d * 16 + j];
    de *= invN;
  }
  bias[j] = pb[j] + de;
}

// ---------------------------------------------------------------- WMMA GEMM
// C[M,Nc] = act(A16[M,K] @ W16[Nc,K]^T + bias).  K % 32 == 0, Nc % 16 == 0.
// block = 128 threads = 4 waves; wave w -> 16-row tile; blockIdx.y -> 16-col tile.
__global__ void k_wmma_gemm(const _Float16* __restrict__ A, const _Float16* __restrict__ W,
                            float* __restrict__ C, const float* __restrict__ bias,
                            int M, int K, int Nc, int relu) {
  int lane = threadIdx.x & 31;
  int wave = threadIdx.x >> 5;
  int rowTile = (blockIdx.x * 4 + wave) * 16;
  if (rowTile >= M) return;
  int colBase = blockIdx.y * 16;
  int hs = lane >> 4;                       // half-wave select
  int mRow = rowTile + (lane & 15);
  int mc = mRow < M ? mRow : M - 1;         // clamp (stores are guarded)
  const _Float16* arow = A + (long)mc * K + hs * 8;
  int nCol = colBase + (lane & 15);
  const _Float16* wrow = W + (long)nCol * K + hs * 8;
  v8f acc = {};
  for (int k0 = 0; k0 < K; k0 += 32) {
    AV16 a, b;
    a.h[0] = *(const v8h*)(arow + k0);
    a.h[1] = *(const v8h*)(arow + k0 + 16);
    b.h[0] = *(const v8h*)(wrow + k0);
    b.h[1] = *(const v8h*)(wrow + k0 + 16);
    acc = __builtin_amdgcn_wmma_f32_16x16x32_f16(false, a.v, false, b.v,
                                                 (short)0, acc, false, false);
  }
  float badd = bias ? bias[nCol] : 0.f;
#pragma unroll
  for (int v = 0; v < 8; ++v) {
    int m = rowTile + v + hs * 8;
    if (m < M) {
      float r = acc[v] + badd;
      if (relu) r = fmaxf(r, 0.f);
      C[(long)m * Nc + nCol] = r;
    }
  }
}

// --------------------------- edge-encoder GEMM: A row e = [h16[row]|h16[col]|ea16[e]]
DEVINL v8h efseg(const _Float16* hr, const _Float16* hc, const _Float16* ep, int k) {
  if (k < 64)  return *(const v8h*)(hr + k);
  if (k < 128) return *(const v8h*)(hc + (k - 64));
  if (k < 144) return *(const v8h*)(ep + (k - 128));
  v8h z = {};
  return z;
}

__global__ void k_wmma_edge_gemm(const _Float16* __restrict__ H16, const _Float16* __restrict__ EA16,
                                 const long long* __restrict__ ei, const _Float16* __restrict__ W,
                                 const float* __restrict__ bias, float* __restrict__ C, long E) {
  const int K = 144, Nc = 64;
  int lane = threadIdx.x & 31;
  int wave = threadIdx.x >> 5;
  long rowTile = ((long)blockIdx.x * 4 + wave) * 16;
  if (rowTile >= E) return;
  int colBase = blockIdx.y * 16;
  int hs = lane >> 4;
  long e = rowTile + (lane & 15);
  if (e >= E) e = E - 1;
  long r = ei[e], c = ei[E + e];
  const _Float16* hr = H16 + r * 64;
  const _Float16* hc = H16 + c * 64;
  const _Float16* ep = EA16 + e * 16;
  int nCol = colBase + (lane & 15);
  const _Float16* wrow = W + (long)nCol * K + hs * 8;
  int aoff = hs * 8;
  v8f acc = {};
  v8h z = {};
#pragma unroll
  for (int k0 = 0; k0 < K; k0 += 32) {      // k0 = 0,32,64,96,128 (tail zero-padded)
    AV16 a, b;
    a.h[0] = efseg(hr, hc, ep, k0 + aoff);
    a.h[1] = efseg(hr, hc, ep, k0 + 16 + aoff);
    int kb = k0 + hs * 8;
    b.h[0] = (kb < K)      ? *(const v8h*)(wrow + k0)      : z;
    b.h[1] = (kb + 16 < K) ? *(const v8h*)(wrow + k0 + 16) : z;
    acc = __builtin_amdgcn_wmma_f32_16x16x32_f16(false, a.v, false, b.v,
                                                 (short)0, acc, false, false);
  }
  float badd = bias[nCol];
#pragma unroll
  for (int v = 0; v < 8; ++v) {
    long m = rowTile + v + hs * 8;
    if (m < E) C[m * Nc + nCol] = fmaxf(acc[v] + badd, 0.f);
  }
}

// ---------------------------------------------------------------- GAT pieces
__global__ void k_att_node(const float* __restrict__ hf, const float* __restrict__ as,
                           const float* __restrict__ ad, float* __restrict__ oas,
                           float* __restrict__ oad, long N, int heads) {
  long i = (long)blockIdx.x * blockDim.x + threadIdx.x;
  if (i >= N * heads) return;
  int h = (int)(i % heads);
  const float* f = hf + i * 64;
  float s0 = 0.f, s1 = 0.f;
#pragma unroll 8
  for (int cc = 0; cc < 64; ++cc) {
    float v = f[cc];
    s0 += v * as[h * 64 + cc];
    s1 += v * ad[h * 64 + cc];
  }
  oas[i] = s0;
  oad[i] = s1;
}

// attE[h][d] = sum_c att_edge[h,c] * we[h*64+c, d]   (collapse edge transform)
__global__ void k_att_eproj(const float* __restrict__ ae, const float* __restrict__ we,
                            float* __restrict__ attE, int heads) {
  int i = threadIdx.x;
  if (i >= heads * 16) return;
  int h = i >> 4, d = i & 15;
  float s = 0.f;
  for (int cc = 0; cc < 64; ++cc) s += ae[h * 64 + cc] * we[(h * 64 + cc) * 16 + d];
  attE[i] = s;
}

__global__ void k_edge_logit(const long long* __restrict__ ei, const float* __restrict__ ea,
                             const float* __restrict__ as, const float* __restrict__ ad,
                             const float* __restrict__ attE, float* __restrict__ lg,
                             float* __restrict__ mmax, long E, int heads) {
  long e = (long)blockIdx.x * blockDim.x + threadIdx.x;
  if (e >= E) return;
  long r = ei[e], c = ei[E + e];
  float av[16];
#pragma unroll
  for (int d = 0; d < 16; ++d) av[d] = ea[e * 16 + d];
  for (int h = 0; h < heads; ++h) {
    float s = as[r * heads + h] + ad[c * heads + h];
#pragma unroll
    for (int d = 0; d < 16; ++d) s += av[d] * attE[h * 16 + d];
    float l = lrelu02(s);
    lg[e * heads + h] = l;
    atomicMaxF(&mmax[c * heads + h], l);
  }
}

__global__ void k_loop_logit(const float* __restrict__ la, const float* __restrict__ as,
                             const float* __restrict__ ad, const float* __restrict__ attE,
                             float* __restrict__ exl, float* __restrict__ mmax, long N, int heads) {
  long n = (long)blockIdx.x * blockDim.x + threadIdx.x;
  if (n >= N) return;
  float av[16];
#pragma unroll
  for (int d = 0; d < 16; ++d) av[d] = la[n * 16 + d];
  for (int h = 0; h < heads; ++h) {
    float s = as[n * heads + h] + ad[n * heads + h];
#pragma unroll
    for (int d = 0; d < 16; ++d) s += av[d] * attE[h * 16 + d];
    float l = lrelu02(s);
    exl[n * heads + h] = l;
    atomicMaxF(&mmax[n * heads + h], l);
  }
}

__global__ void k_edge_exp(const long long* __restrict__ ei, float* __restrict__ lg,
                           const float* __restrict__ mmax, float* __restrict__ den,
                           long E, int heads) {
  long e = (long)blockIdx.x * blockDim.x + threadIdx.x;
  if (e >= E) return;
  long c = ei[E + e];
  for (int h = 0; h < heads; ++h) {
    float ex = __expf(lg[e * heads + h] - mmax[c * heads + h]);
    lg[e * heads + h] = ex;
    atomicAdd(&den[c * heads + h], ex);
  }
}

__global__ void k_loop_exp(float* __restrict__ exl, const float* __restrict__ mmax,
                           float* __restrict__ den, long N, int heads) {
  long n = (long)blockIdx.x * blockDim.x + threadIdx.x;
  if (n >= N) return;
  for (int h = 0; h < heads; ++h) {
    float ex = __expf(exl[n * heads + h] - mmax[n * heads + h]);
    exl[n * heads + h] = ex;
    atomicAdd(&den[n * heads + h], ex);
  }
}

__global__ void k_edge_agg(const long long* __restrict__ ei, const float* __restrict__ lg,
                           const float* __restrict__ den, const float* __restrict__ hf,
                           float* __restrict__ agg, long E, int heads) {
  long i = (long)blockIdx.x * blockDim.x + threadIdx.x;
  long tot = E * heads * 16;
  if (i >= tot) return;
  int g = (int)(i & 15);
  long t = i >> 4;
  int h = (int)(t % heads);
  long e = t / heads;
  long r = ei[e], c = ei[E + e];
  float alpha = lg[e * heads + h] / (den[c * heads + h] + 1e-16f);
  const float4 v = *(const float4*)(hf + (r * heads + h) * 64 + g * 4);
  float* dst = agg + (c * heads + h) * 64 + g * 4;
  atomicAdd(dst + 0, alpha * v.x);
  atomicAdd(dst + 1, alpha * v.y);
  atomicAdd(dst + 2, alpha * v.z);
  atomicAdd(dst + 3, alpha * v.w);
}

__global__ void k_loop_agg(const float* __restrict__ exl, const float* __restrict__ den,
                           const float* __restrict__ hf, float* __restrict__ agg,
                           long N, int heads) {
  long i = (long)blockIdx.x * blockDim.x + threadIdx.x;
  if (i >= N * heads * 16) return;
  int g = (int)(i & 15);
  long t = i >> 4;
  int h = (int)(t % heads);
  long n = t / heads;
  float alpha = exl[n * heads + h] / (den[n * heads + h] + 1e-16f);
  const float4 v = *(const float4*)(hf + (n * heads + h) * 64 + g * 4);
  float* dst = agg + (n * heads + h) * 64 + g * 4;   // unique slot per thread here
  dst[0] += alpha * v.x;
  dst[1] += alpha * v.y;
  dst[2] += alpha * v.z;
  dst[3] += alpha * v.w;
}

__global__ void k_conv_epi(const float* __restrict__ agg, const float* __restrict__ b,
                           float* __restrict__ out, long N, int HC, int elu) {
  long i = (long)blockIdx.x * blockDim.x + threadIdx.x;
  if (i >= N * (long)HC) return;
  int j = (int)(i % HC);
  float v = agg[i] + b[j];
  if (elu) v = v > 0.f ? v : (__expf(v) - 1.f);
  out[i] = v;
}

// ------------------------------------------ two-layer sigmoid head: [M,64]->[M]
__global__ void k_head(const float* __restrict__ F, const float* __restrict__ w1,
                       const float* __restrict__ b1, const float* __restrict__ w2,
                       const float* __restrict__ b2, float* __restrict__ out, long M) {
  __shared__ float sw1[2048];
  __shared__ float sb1[32];
  __shared__ float sw2[32];
  __shared__ float sb2[1];
  for (int t = threadIdx.x; t < 2048; t += blockDim.x) sw1[t] = w1[t];
  if (threadIdx.x < 32) { sb1[threadIdx.x] = b1[threadIdx.x]; sw2[threadIdx.x] = w2[threadIdx.x]; }
  if (threadIdx.x == 0) sb2[0] = b2[0];
  __syncthreads();
  long m = (long)blockIdx.x * blockDim.x + threadIdx.x;
  if (m >= M) return;
  float f[64];
  const float* src = F + m * 64;
#pragma unroll
  for (int cc = 0; cc < 64; ++cc) f[cc] = src[cc];
  float s = sb2[0];
  for (int j = 0; j < 32; ++j) {
    float zacc = sb1[j];
    const float* wr = &sw1[j * 64];
#pragma unroll
    for (int cc = 0; cc < 64; ++cc) zacc += wr[cc] * f[cc];
    s += sw2[j] * fmaxf(zacc, 0.f);
  }
  out[m] = 1.f / (1.f + __expf(-s));
}

// ================================================================ host
extern "C" void kernel_launch(void* const* d_in, const int* in_sizes, int n_in,
                              void* d_out, int out_size, void* d_ws, size_t ws_size,
                              hipStream_t stream) {
  (void)n_in; (void)out_size; (void)ws_size;
  const long N = in_sizes[0] / 128;
  const long E = in_sizes[2] / 16;

  const float* x        = (const float*)d_in[0];
  const long long* ei   = (const long long*)d_in[1];
  const float* ea       = (const float*)d_in[2];
  const long long* dom  = (const long long*)d_in[3];

  int pi = 4;
  auto nextP = [&]() -> const float* { return (const float*)d_in[pi++]; };
  const float* dom_emb = nextP();
  const float* proj_w  = nextP();
  const float* proj_b  = nextP();
  struct Gat { const float *w, *we, *as, *ad, *ae, *b; };
  auto nextG = [&]() -> Gat {
    Gat g; g.w = nextP(); g.we = nextP(); g.as = nextP();
    g.ad = nextP(); g.ae = nextP(); g.b = nextP(); return g;
  };
  Gat c1 = nextG(); Gat c2 = nextG(); Gat c3 = nextG(); Gat c4 = nextG();
  const float* enc_w = nextP();
  const float* enc_b = nextP();
  struct Mlp { const float *w1, *b1, *w2, *b2; };
  auto nextM = [&]() -> Mlp {
    Mlp m; m.w1 = nextP(); m.b1 = nextP(); m.w2 = nextP(); m.b2 = nextP(); return m;
  };
  Mlp imp = nextM(); Mlp rel = nextM();
  const float* ee_w = nextP();
  const float* ee_b = nextP();
  Mlp conf = nextM(); Mlp sev = nextM(); Mlp prob = nextM();

  // ---------------- workspace carve
  char* wsb = (char*)d_ws;
  size_t off = 0;
  auto carve = [&](size_t bytes) -> void* {
    void* p = wsb + off;
    off = (off + bytes + 255) & ~(size_t)255;
    return p;
  };
  float*    hA    = (float*)carve((size_t)N * 256 * 4);   // activations (ping)
  float*    hB    = (float*)carve((size_t)N * 256 * 4);   // hfeat / nf
  float*    agg   = (float*)carve((size_t)N * 256 * 4);   // aggregation accum
  _Float16* x16   = (_Float16*)carve((size_t)N * 256 * 2);
  _Float16* w16   = (_Float16*)carve((size_t)256 * 256 * 2);
  _Float16* ea16  = (_Float16*)carve((size_t)E * 16 * 2);
  float*    asrc  = (float*)carve((size_t)N * 4 * 4);
  float*    adst  = (float*)carve((size_t)N * 4 * 4);
  float*    mmax  = (float*)carve((size_t)N * 4 * 4);
  float*    den   = (float*)carve((size_t)N * 4 * 4);
  float*    exl   = (float*)carve((size_t)N * 4 * 4);
  float*    lg    = (float*)carve((size_t)E * 4 * 4);
  float*    deg   = (float*)carve((size_t)N * 4);
  float*    lattr = (float*)carve((size_t)N * 16 * 4);
  float*    attE  = (float*)carve(64 * 4);
  float*    dcnt  = (float*)carve(4 * 4);
  float*    pbias = (float*)carve(256 * 4);
  float*    ef    = (float*)carve((size_t)E * 64 * 4);    // edge features

  auto g1 = [](long n) { return dim3((unsigned)((n + 255) / 256)); };

  // ---------------- preprocessing (once)
  k_fill<<<g1(N), 256, 0, stream>>>(deg, 0.f, N);
  k_fill<<<g1(N * 16), 256, 0, stream>>>(lattr, 0.f, N * 16);
  k_fill<<<1, 256, 0, stream>>>(dcnt, 0.f, 4);
  k_deg_loop<<<g1(E), 256, 0, stream>>>(ei, ea, deg, lattr, E);
  k_loop_fin<<<g1(N * 16), 256, 0, stream>>>(deg, lattr, N);
  k_domcnt<<<g1(N), 256, 0, stream>>>(dom, dcnt, N);
  k_projbias<<<1, 64, 0, stream>>>(dcnt, dom_emb, proj_b, pbias, 1.f / (float)N);
  k_cvt16<<<g1(E * 16), 256, 0, stream>>>(ea, ea16, E * 16);

  // ---------------- projection: hA = x @ proj_w^T + (proj_b + pad(de))
  k_cvt16<<<g1(N * 128), 256, 0, stream>>>(x, x16, N * 128);
  k_cvt16<<<g1(64 * 128), 256, 0, stream>>>(proj_w, w16, 64 * 128);
  {
    dim3 gg((unsigned)((N + 63) / 64), 4);
    k_wmma_gemm<<<gg, 128, 0, stream>>>(x16, w16, hA, pbias, (int)N, 128, 64, 0);
  }

  // ---------------- GAT convolutions
  auto run_conv = [&](int Fin, int heads, const Gat& g, bool elu) {
    int HC = heads * 64;
    k_cvt16<<<g1(N * Fin), 256, 0, stream>>>(hA, x16, N * (long)Fin);
    k_cvt16<<<g1((long)HC * Fin), 256, 0, stream>>>(g.w, w16, (long)HC * Fin);
    dim3 gg((unsigned)((N + 63) / 64), (unsigned)(HC / 16));
    k_wmma_gemm<<<gg, 128, 0, stream>>>(x16, w16, hB, nullptr, (int)N, Fin, HC, 0);
    k_att_node<<<g1(N * heads), 256, 0, stream>>>(hB, g.as, g.ad, asrc, adst, N, heads);
    k_att_eproj<<<1, 64, 0, stream>>>(g.ae, g.we, attE, heads);
    k_fill<<<g1(N * heads), 256, 0, stream>>>(mmax, -3.0e38f, N * heads);
    k_fill<<<g1(N * heads), 256, 0, stream>>>(den, 0.f, N * heads);
    k_edge_logit<<<g1(E), 256, 0, stream>>>(ei, ea, asrc, adst, attE, lg, mmax, E, heads);
    k_loop_logit<<<g1(N), 256, 0, stream>>>(lattr, asrc, adst, attE, exl, mmax, N, heads);
    k_edge_exp<<<g1(E), 256, 0, stream>>>(ei, lg, mmax, den, E, heads);
    k_loop_exp<<<g1(N), 256, 0, stream>>>(exl, mmax, den, N, heads);
    k_fill<<<g1(N * (long)HC), 256, 0, stream>>>(agg, 0.f, N * (long)HC);
    k_edge_agg<<<g1(E * heads * 16), 256, 0, stream>>>(ei, lg, den, hB, agg, E, heads);
    k_loop_agg<<<g1(N * heads * 16), 256, 0, stream>>>(exl, den, hB, agg, N, heads);
    k_conv_epi<<<g1(N * (long)HC), 256, 0, stream>>>(agg, g.b, hA, N, HC, elu ? 1 : 0);
  };
  run_conv(64,  4, c1, true);
  run_conv(256, 4, c2, true);
  run_conv(256, 2, c3, true);
  run_conv(128, 1, c4, false);           // heads=1, mean==identity, no ELU

  // ---------------- node heads: nf = relu(h @ enc_w^T + enc_b)
  float* dout = (float*)d_out;
  k_cvt16<<<g1(N * 64), 256, 0, stream>>>(hA, x16, N * 64);   // final h in f16 (also edge GEMM input)
  k_cvt16<<<g1(64 * 64), 256, 0, stream>>>(enc_w, w16, 64 * 64);
  {
    dim3 gg((unsigned)((N + 63) / 64), 4);
    k_wmma_gemm<<<gg, 128, 0, stream>>>(x16, w16, hB, enc_b, (int)N, 64, 64, 1);
  }
  k_head<<<g1(N), 256, 0, stream>>>(hB, imp.w1, imp.b1, imp.w2, imp.b2, dout, N);
  k_head<<<g1(N), 256, 0, stream>>>(hB, rel.w1, rel.b1, rel.w2, rel.b2, dout + N, N);

  // ---------------- edge heads: ef = relu([h[row]|h[col]|ea] @ ee_w^T + ee_b)
  k_cvt16<<<g1(64 * 144), 256, 0, stream>>>(ee_w, w16, 64 * 144);
  {
    dim3 gg((unsigned)((E + 63) / 64), 4);
    k_wmma_edge_gemm<<<gg, 128, 0, stream>>>(x16, ea16, ei, w16, ee_b, ef, E);
  }
  k_head<<<g1(E), 256, 0, stream>>>(ef, conf.w1, conf.b1, conf.w2, conf.b2, dout + 2 * N, E);
  k_head<<<g1(E), 256, 0, stream>>>(ef, sev.w1,  sev.b1,  sev.w2,  sev.b2,  dout + 2 * N + E, E);
  k_head<<<g1(E), 256, 0, stream>>>(ef, prob.w1, prob.b1, prob.w2, prob.b2, dout + 2 * N + 2 * E, E);
}
